// X_BERT_VQVAE_38723425140846
// MI455X (gfx1250) — matrix operand order repaired and checked by reading, not compiled, not run on hardware
//
#include <hip/hip_runtime.h>

typedef __bf16 bf16_t;
typedef __attribute__((ext_vector_type(8)))  __bf16 v8bf;
typedef __attribute__((ext_vector_type(16))) __bf16 v16bf;
typedef __attribute__((ext_vector_type(8)))  float  v8f;

#define Bc 64
#define Tc 276
#define Dc 256
#define Kc 8192
#define Nc (Bc*Tc)                      /* 17664 rows */
#define ROWTILES (Nc/16)                /* 1104 */
#define WAVES_PER_BLK 8
#define MAIN_BLOCKS (ROWTILES/WAVES_PER_BLK) /* 138 */
#define KTILES (Kc/16)                  /* 512 */

// ---------------------------------------------------------------------------
// Split fp32 -> bf16 hi/lo pair; also compute ||c_k||^2 for the codebook.
// One wave32 per row.
// ---------------------------------------------------------------------------
__global__ void __launch_bounds__(256) prep_codebook_kernel(
    const float* __restrict__ cb, bf16_t* __restrict__ hi,
    bf16_t* __restrict__ lo, float* __restrict__ cnorm)
{
  int gwave = (blockIdx.x * blockDim.x + threadIdx.x) >> 5;
  int lane  = threadIdx.x & 31;
  if (gwave >= Kc) return;
  const float* row = cb + (size_t)gwave * Dc;
  float s = 0.f;
  #pragma unroll
  for (int i = 0; i < Dc/32; ++i) {
    int e = lane + i*32;
    float x = row[e];
    bf16_t h = (bf16_t)x;
    bf16_t l = (bf16_t)(x - (float)h);
    hi[(size_t)gwave*Dc + e] = h;
    lo[(size_t)gwave*Dc + e] = l;
    s += x * x;
  }
  #pragma unroll
  for (int m = 16; m >= 1; m >>= 1) s += __shfl_xor(s, m, 32);
  if (lane == 0) cnorm[gwave] = s;
}

__global__ void __launch_bounds__(256) prep_z_kernel(
    const float* __restrict__ z, bf16_t* __restrict__ hi, bf16_t* __restrict__ lo)
{
  int gwave = (blockIdx.x * blockDim.x + threadIdx.x) >> 5;
  int lane  = threadIdx.x & 31;
  if (gwave >= Nc) return;
  const float* row = z + (size_t)gwave * Dc;
  #pragma unroll
  for (int i = 0; i < Dc/32; ++i) {
    int e = lane + i*32;
    float x = row[e];
    bf16_t h = (bf16_t)x;
    bf16_t l = (bf16_t)(x - (float)h);
    hi[(size_t)gwave*Dc + e] = h;
    lo[(size_t)gwave*Dc + e] = l;
  }
}

// ---------------------------------------------------------------------------
// Async-copy one 16-code tile (hi 8KB + lo 8KB) from global into LDS using
// GLOBAL_LOAD_ASYNC_TO_LDS_B128 (ASYNCcnt path, bypasses VGPRs).
// Each of 256 threads moves 2x16B from hi and 2x16B from lo.
// ---------------------------------------------------------------------------
__device__ __forceinline__ void stage_tile_async(
    const bf16_t* __restrict__ gh, const bf16_t* __restrict__ gl,
    bf16_t* lds_h, bf16_t* lds_l, int tid)
{
  unsigned ldsh = (unsigned)(size_t)lds_h;   // generic->LDS offset: addr[31:0]
  unsigned ldsl = (unsigned)(size_t)lds_l;
  #pragma unroll
  for (int p = 0; p < 2; ++p) {
    unsigned off = (unsigned)tid * 16u + (unsigned)p * 4096u;
    unsigned dh  = ldsh + off;
    unsigned dl  = ldsl + off;
    asm volatile("global_load_async_to_lds_b128 %0, %1, %2"
                 :: "v"(dh), "v"(off), "s"(gh) : "memory");
    asm volatile("global_load_async_to_lds_b128 %0, %1, %2"
                 :: "v"(dl), "v"(off), "s"(gl) : "memory");
  }
}

__device__ __forceinline__ void wait_async_zero()
{
  asm volatile("s_wait_asynccnt 0x0" ::: "memory");
}

// ---------------------------------------------------------------------------
// Main fused kernel: scores via bf16x3 WMMA, running argmin, gather, loss.
// Block = 256 threads = 8 waves; wave w handles rows [blk*128 + w*16, +16).
// Codebook tiles (16 codes, hi+lo) double-buffered in LDS, shared by 8 waves.
// ---------------------------------------------------------------------------
__global__ void __launch_bounds__(256, 1) vq_argmin_kernel(
    const float* __restrict__ z,  const float* __restrict__ cb,
    const bf16_t* __restrict__ zhi, const bf16_t* __restrict__ zlo,
    const bf16_t* __restrict__ cbhi, const bf16_t* __restrict__ cblo,
    const float* __restrict__ cnorm,
    float* __restrict__ zq_out, float* __restrict__ idx_out,
    float* __restrict__ partial)
{
  __shared__ __align__(16) bf16_t sB[2][2][16 * Dc];   // [buf][hi/lo][16 codes x 256]
  __shared__ float sCn[2][16];

  const int lane    = threadIdx.x & 31;
  const int wave    = threadIdx.x >> 5;
  const int rowtile = blockIdx.x * WAVES_PER_BLK + wave;
  const int row0    = rowtile * 16;

  // -------- load this wave's A tile (16 z-rows, hi+lo) into registers.
  // A layout (16-bit 16x32): lane L<16 -> row L, K = {c*32+0..7, c*32+16..23};
  // lane L+16 -> row L, K = {c*32+8..15, c*32+24..31}.
  const int arow  = row0 + (lane & 15);
  const int akoff = (lane >> 4) * 8;
  v16bf Ahi[8], Alo[8];
  #pragma unroll
  for (int c = 0; c < 8; ++c) {
    const bf16_t* ph = zhi + (size_t)arow * Dc + c * 32 + akoff;
    v8bf h0 = *(const v8bf*)ph;
    v8bf h1 = *(const v8bf*)(ph + 16);
    Ahi[c] = __builtin_shufflevector(h0, h1, 0,1,2,3,4,5,6,7,8,9,10,11,12,13,14,15);
    const bf16_t* pl = zlo + (size_t)arow * Dc + c * 32 + akoff;
    v8bf l0 = *(const v8bf*)pl;
    v8bf l1 = *(const v8bf*)(pl + 16);
    Alo[c] = __builtin_shufflevector(l0, l1, 0,1,2,3,4,5,6,7,8,9,10,11,12,13,14,15);
  }

  float best[8];
  int   bidx[8];
  #pragma unroll
  for (int v = 0; v < 8; ++v) { best[v] = 3.0e38f; bidx[v] = 0; }

  // -------- stage code-tile 0 into buffer 0 (async -> LDS)
  stage_tile_async(cbhi, cblo, &sB[0][0][0], &sB[0][1][0], threadIdx.x);
  if (threadIdx.x < 16) sCn[0][threadIdx.x] = cnorm[threadIdx.x];
  wait_async_zero();
  __syncthreads();

  for (int kt = 0; kt < KTILES; ++kt) {
    const int buf = kt & 1;

    // stage next tile into the other buffer (async, overlaps with WMMA below)
    if (kt + 1 < KTILES) {
      stage_tile_async(cbhi + (size_t)(kt + 1) * 16 * Dc,
                       cblo + (size_t)(kt + 1) * 16 * Dc,
                       &sB[buf ^ 1][0][0], &sB[buf ^ 1][1][0], threadIdx.x);
      if (threadIdx.x < 16) sCn[buf ^ 1][threadIdx.x] = cnorm[(kt + 1) * 16 + threadIdx.x];
    }
    // L2 prefetch of the tile after next (global_prefetch_b8)
    if (kt + 2 < KTILES) {
      __builtin_prefetch(cbhi + (size_t)(kt + 2) * 16 * Dc + threadIdx.x * 16, 0, 0);
      __builtin_prefetch(cblo + (size_t)(kt + 2) * 16 * Dc + threadIdx.x * 16, 0, 0);
    }

    // B layout (16-bit 32x16): lane n<16 -> code n, K = {c*32+0..7, c*32+16..23};
    // lane n+16 -> code n, K = {c*32+8..15, c*32+24..31}.
    const int brow  = lane & 15;
    const int bkoff = (lane >> 4) * 8;
    // Two independent accumulation chains for WMMA-pipe ILP:
    //   acc0 = sum_c z_hi.c_hi      acc1 = sum_c (z_hi.c_lo + z_lo.c_hi)
    v8f acc0 = {0.f, 0.f, 0.f, 0.f, 0.f, 0.f, 0.f, 0.f};
    v8f acc1 = {0.f, 0.f, 0.f, 0.f, 0.f, 0.f, 0.f, 0.f};
    #pragma unroll
    for (int c = 0; c < 8; ++c) {
      const bf16_t* ph = &sB[buf][0][brow * Dc + c * 32 + bkoff];
      v8bf h0 = *(const v8bf*)ph;
      v8bf h1 = *(const v8bf*)(ph + 16);
      v16bf Bh = __builtin_shufflevector(h0, h1, 0,1,2,3,4,5,6,7,8,9,10,11,12,13,14,15);
      const bf16_t* pl = &sB[buf][1][brow * Dc + c * 32 + bkoff];
      v8bf l0 = *(const v8bf*)pl;
      v8bf l1 = *(const v8bf*)(pl + 16);
      v16bf Bl = __builtin_shufflevector(l0, l1, 0,1,2,3,4,5,6,7,8,9,10,11,12,13,14,15);
      acc0 = __builtin_amdgcn_wmma_f32_16x16x32_bf16(false, Ahi[c], false, Bh, (short)0, acc0, false, false);
      acc1 = __builtin_amdgcn_wmma_f32_16x16x32_bf16(false, Ahi[c], false, Bl, (short)0, acc1, false, false);
      acc1 = __builtin_amdgcn_wmma_f32_16x16x32_bf16(false, Alo[c], false, Bh, (short)0, acc1, false, false);
    }

    // dist(m,n) = ||c_n||^2 - 2 z_m.c_n   (||z_m||^2 constant per row, dropped)
    const float cn = sCn[buf][lane & 15];
    const int   id = kt * 16 + (lane & 15);
    #pragma unroll
    for (int v = 0; v < 8; ++v) {
      float dist = cn - 2.0f * (acc0[v] + acc1[v]);
      if (dist < best[v]) { best[v] = dist; bidx[v] = id; }   // strict < : first-min wins
    }

    // make this wave's async tile-writes visible, then block-wide barrier
    wait_async_zero();
    __syncthreads();
  }

  // -------- argmin across the 16 lanes holding each row (xor within half-wave)
  #pragma unroll
  for (int v = 0; v < 8; ++v) {
    float d = best[v];
    int   i = bidx[v];
    #pragma unroll
    for (int m = 8; m >= 1; m >>= 1) {
      float od = __shfl_xor(d, m, 32);
      int   oi = __shfl_xor(i, m, 32);
      if (od < d || (od == d && oi < i)) { d = od; i = oi; }  // tie -> smallest index
    }
    best[v] = d;
    bidx[v] = i;
  }

  // write min_idx (as float, per harness output dtype); rows: v + 8*(lane/16)
  if ((lane & 15) == 0) {
    const int g = lane >> 4;
    #pragma unroll
    for (int v = 0; v < 8; ++v)
      idx_out[row0 + v + 8 * g] = (float)bidx[v];
  }

  // -------- gather z_q = codebook[idx] (exact fp32), write straight-through
  // output (forward value == z_q), and accumulate sum((z_q - z)^2).
  float lsum = 0.f;
  #pragma unroll
  for (int v = 0; v < 8; ++v) {
    #pragma unroll
    for (int g = 0; g < 2; ++g) {
      const int idx = __shfl(bidx[v], g * 16, 32);
      const int row = row0 + v + 8 * g;
      const float4* crow = (const float4*)(cb + (size_t)idx * Dc);
      const float4* zrow = (const float4*)(z  + (size_t)row * Dc);
      float4* orow = (float4*)(zq_out + (size_t)row * Dc);
      #pragma unroll
      for (int p = 0; p < 2; ++p) {
        const int e = lane + p * 32;   // 64 float4 per row
        float4 cv = crow[e];
        float4 zv = zrow[e];
        orow[e] = cv;
        float d0 = cv.x - zv.x, d1 = cv.y - zv.y, d2 = cv.z - zv.z, d3 = cv.w - zv.w;
        lsum += d0 * d0 + d1 * d1 + d2 * d2 + d3 * d3;
      }
    }
  }
  #pragma unroll
  for (int m = 16; m >= 1; m >>= 1) lsum += __shfl_xor(lsum, m, 32);
  if (lane == 0) partial[rowtile] = lsum;
}

// ---------------------------------------------------------------------------
// Deterministic final loss reduction (fixed summation order, no float atomics)
// ---------------------------------------------------------------------------
__global__ void finalize_kernel(const float* __restrict__ partial, float* __restrict__ out)
{
  if (threadIdx.x == 0 && blockIdx.x == 0) {
    float s = 0.f;
    for (int i = 0; i < ROWTILES; ++i) s += partial[i];
    const float m = s / (float)((size_t)Nc * Dc);
    out[0] = m;  // loss_vq
    out[1] = m;  // loss_commit (numerically identical)
  }
}

extern "C" void kernel_launch(void* const* d_in, const int* in_sizes, int n_in,
                              void* d_out, int out_size, void* d_ws, size_t ws_size,
                              hipStream_t stream)
{
  const float* z  = (const float*)d_in[0];   // [B,T,D] fp32
  const float* cb = (const float*)d_in[1];   // [K,D]  fp32

  // d_out layout: [loss_vq, loss_commit, z_q (N*D), min_idx (N)] as float
  float* out     = (float*)d_out;
  float* zq_out  = out + 2;
  float* idx_out = out + 2 + (size_t)Nc * Dc;

  // workspace layout (bytes)
  char* ws = (char*)d_ws;
  bf16_t* zhi   = (bf16_t*)(ws);
  bf16_t* zlo   = (bf16_t*)(ws + (size_t)Nc * Dc * 2);
  bf16_t* cbhi  = (bf16_t*)(ws + (size_t)Nc * Dc * 4);
  bf16_t* cblo  = (bf16_t*)(ws + (size_t)Nc * Dc * 4 + (size_t)Kc * Dc * 2);
  float*  cnorm = (float*) (ws + (size_t)Nc * Dc * 4 + (size_t)Kc * Dc * 4);
  float*  part  = (float*) (ws + (size_t)Nc * Dc * 4 + (size_t)Kc * Dc * 4 + (size_t)Kc * 4);

  prep_z_kernel<<<Nc / 8, 256, 0, stream>>>(z, zhi, zlo);
  prep_codebook_kernel<<<Kc / 8, 256, 0, stream>>>(cb, cbhi, cblo, cnorm);
  vq_argmin_kernel<<<MAIN_BLOCKS, 256, 0, stream>>>(z, cb, zhi, zlo, cbhi, cblo,
                                                    cnorm, zq_out, idx_out, part);
  finalize_kernel<<<1, 32, 0, stream>>>(part, out);
}
